// ScaledDotProductAttention_39651138077575
// MI455X (gfx1250) — matrix operand order repaired and checked by reading, not compiled, not run on hardware
//
#include <hip/hip_runtime.h>
#include <hip/hip_bf16.h>
#include <math.h>

// Shapes fixed by the reference: B=4, H=16, L=1024, D=64.
#define BB 4
#define HH 16
#define LL 1024
#define DD 64

typedef __attribute__((ext_vector_type(16))) __bf16 v16bf;
typedef __attribute__((ext_vector_type(4)))  __bf16 v4bf;
typedef __attribute__((ext_vector_type(8)))  float  v8f;
typedef __attribute__((ext_vector_type(4)))  float  v4f;

__device__ __forceinline__ __bf16 f2bf(float x) { return (__bf16)x; }

#define WMMA_BF16(a, b, c) \
  __builtin_amdgcn_wmma_f32_16x16x32_bf16(false, (a), false, (b), (short)0, (c), false, false)

// fast 1/(x*x) with x==0 -> 1.0 (reference's transformed bias)
__device__ __forceinline__ float bias_xform(float x) {
    return (x == 0.0f) ? 1.0f : __builtin_amdgcn_rcpf(x * x);
}

// ---------------------------------------------------------------------------
// Loss kernel: one block per (b, q-tile of 16, k-tile of 64). Loops over all
// H heads, recomputes the masked score tile via WMMA, accumulates head-sums of
// score and transformed bias per lane (each lane owns unique C elements),
// then block-reduces the squared diff of head-means into a global accumulator.
// ---------------------------------------------------------------------------
__global__ __launch_bounds__(128) void sdpa_loss_kernel(
    const float* __restrict__ Q, const float* __restrict__ K,
    const int* __restrict__ mask, const float* __restrict__ bias,
    float* __restrict__ acc_out)
{
    __shared__ float  qs[16 * DD];        // Q tile, f32        (4KB)
    __shared__ __bf16 ks[64 * DD];        // K tile, bf16       (8KB)
    __shared__ float  bs[16 * 64];        // transformed bias   (4KB)
    __shared__ float  red[4];

    const int kt  = blockIdx.x;           // 0..15  (64-wide k tile)
    const int qt  = blockIdx.y;           // 0..63  (16-wide q tile)
    const int b   = blockIdx.z;
    const int tid = threadIdx.x;
    const int wv  = tid >> 5;             // wave 0..3 -> 16-col N subtile
    const int ln  = tid & 31;

    const int q0 = qt * 16, k0 = kt * 64;
    const int am   = ln & 15;             // A-matrix row for this lane
    const int kb   = (ln >> 4) << 3;      // unused; A loads use koff (0 or 16)
    const int bn   = ln & 15;             // B-matrix N for this lane
    const int koff = (ln >> 4) << 4;      // B K base (0 or 16)
    const int cm   = (ln >> 4) << 3;      // C row base (0 or 8)

    float sumS[8], sumB[8];
#pragma unroll
    for (int r = 0; r < 8; ++r) { sumS[r] = 0.f; sumB[r] = 0.f; }

    for (int h = 0; h < HH; ++h) {
        __syncthreads();  // protect staging from previous iteration's readers
        const float* qg = Q + (((size_t)b * HH + h) * LL + q0) * DD;
        const float* kg = K + (((size_t)b * HH + h) * LL + k0) * DD;
        const float* bg = bias + (((size_t)b * HH + h) * LL + q0) * LL + k0;

        // wide staging: Q tile (2 x b128/thread), K tile (8 x b128/thread)
        const v4f* qg4 = (const v4f*)qg;
        for (int i = tid; i < 16 * DD / 4; i += 128) ((v4f*)qs)[i] = qg4[i];
        const v4f* kg4 = (const v4f*)kg;
        for (int i = tid; i < 64 * DD / 4; i += 128) {
            v4f v = kg4[i];
            v4bf w; w[0] = f2bf(v.x); w[1] = f2bf(v.y); w[2] = f2bf(v.z); w[3] = f2bf(v.w);
            ((v4bf*)ks)[i] = w;
        }
        // bias tile: non-temporal b128 stream + transform into LDS
        for (int i = tid; i < 256; i += 128) {
            const int row = i >> 4, c4 = i & 15;
            v4f v = __builtin_nontemporal_load((const v4f*)(bg + row * LL) + c4);
            v4f t;
            t.x = bias_xform(v.x); t.y = bias_xform(v.y);
            t.z = bias_xform(v.z); t.w = bias_xform(v.w);
            ((v4f*)bs)[i] = t;
        }
        if (h + 1 < HH) __builtin_prefetch(bg + (size_t)LL * LL, 0, 0);
        __syncthreads();

        v16bf a0, a1, b0, b1;
        const int krow = wv * 16 + bn;
#pragma unroll
        for (int e = 0; e < 16; ++e) {
            const int k = koff + e;                   // 0-15 or 16-31
            a0[e] = f2bf(qs[am * DD + k]);
            a1[e] = f2bf(qs[am * DD + 32 + k]);
            b0[e] = ks[krow * DD + koff + e];
            b1[e] = ks[krow * DD + 32 + koff + e];
        }
        v8f c = {};
        c = WMMA_BF16(a0, b0, c);
        c = WMMA_BF16(a1, b1, c);

#pragma unroll
        for (int r = 0; r < 8; ++r) {
            const int mm   = r + cm;
            const int ncol = wv * 16 + (ln & 15);
            sumS[r] += c[r] * 0.125f;                 // 1/sqrt(64)
            sumB[r] += bs[mm * 64 + ncol];
        }
    }

    // Mask is head-invariant -> masked head-mean is exactly -10000.
    float part = 0.f;
#pragma unroll
    for (int r = 0; r < 8; ++r) {
        const int mm   = r + cm;
        const int ncol = wv * 16 + (ln & 15);
        const int q = q0 + mm, k = k0 + ncol;
        const int mk = mask[((size_t)b * LL + q) * LL + k];
        const float meanS = mk ? (sumS[r] * (1.0f / 16.0f)) : -10000.0f;
        const float meanB = sumB[r] * (1.0f / 16.0f);
        const float d = meanS - meanB;
        part += d * d;
    }
    for (int off = 16; off > 0; off >>= 1) part += __shfl_xor(part, off, 32);
    if (ln == 0) red[wv] = part;
    __syncthreads();
    if (tid == 0) atomicAdd(acc_out, red[0] + red[1] + red[2] + red[3]);
}

// ---------------------------------------------------------------------------
// Attention kernel: one block (4 waves) per (b, h, q-tile of 16).
// Phase 1: S[16][1024] = mask(QK^T/8) in LDS (f32), WMMA bf16.
// Phase 2: row softmax; unnormalized exp kept in S, row-sums saved.
// Phase 3: O = exp(S) @ V via WMMA; row scale 1/rowsum folded into store.
// ---------------------------------------------------------------------------
__global__ __launch_bounds__(128) void sdpa_attn_kernel(
    const float* __restrict__ Q, const float* __restrict__ K,
    const float* __restrict__ V, const int* __restrict__ mask,
    float* __restrict__ out)
{
    extern __shared__ char smem[];
    float*  S    = (float*)smem;                                   // 16*1024 f32 = 64KB
    __bf16* kv   = (__bf16*)(smem + 16 * LL * 4);                  // 64*64 bf16  =  8KB
    int*    ms   = (int*)(smem + 16 * LL * 4 + 64 * DD * 2);       // 16*64 int   =  4KB
    float*  red  = (float*)(smem + 16 * LL * 4 + 64 * DD * 2 + 16 * 64 * 4);
    float*  rsum = red + 16 * 8;                                   // 16 f32

    const int qt  = blockIdx.x;   // 0..63
    const int h   = blockIdx.y;
    const int b   = blockIdx.z;
    const int tid = threadIdx.x;
    const int wv  = tid >> 5;
    const int ln  = tid & 31;

    const int q0   = qt * 16;
    const int am   = ln & 15;
    const int kb   = (ln >> 4) << 3;
    const int bn   = ln & 15;
    const int koff = (ln >> 4) << 4;
    const int cm   = (ln >> 4) << 3;

    const size_t headoff = ((size_t)b * HH + h) * LL;

    // Q A-fragments straight from global (one tile per block, reused 16x).
    const float* qg = Q + (headoff + q0) * DD;
    v16bf aq0, aq1;
#pragma unroll
    for (int e = 0; e < 16; ++e) {
        const int k = koff + e;
        aq0[e] = f2bf(qg[am * DD + k]);
        aq1[e] = f2bf(qg[am * DD + 32 + k]);
    }

    // ---- Phase 1: masked, scaled scores into LDS ----
    const int* mbase = mask + ((size_t)b * LL + q0) * LL;
    for (int kt = 0; kt < LL / 64; ++kt) {
        __syncthreads();
        const float* kg = K + (headoff + kt * 64) * DD;
        const v4f* kg4 = (const v4f*)kg;
        for (int i = tid; i < 64 * DD / 4; i += 128) {
            v4f v = kg4[i];
            v4bf w; w[0] = f2bf(v.x); w[1] = f2bf(v.y); w[2] = f2bf(v.z); w[3] = f2bf(v.w);
            ((v4bf*)kv)[i] = w;
        }
        const int* mrow = mbase + kt * 64;
        for (int i = tid; i < 256; i += 128) {
            const int row = i >> 4, c4 = i & 15;
            ((int4*)ms)[i] = ((const int4*)(mrow + row * LL))[c4];
        }
        if (kt + 1 < LL / 64) __builtin_prefetch(kg + 64 * DD, 0, 0);
        __syncthreads();

        v16bf b0, b1;
        const int krow = wv * 16 + bn;
#pragma unroll
        for (int e = 0; e < 16; ++e) {
            b0[e] = kv[krow * DD + koff + e];
            b1[e] = kv[krow * DD + 32 + koff + e];
        }
        v8f c = {};
        c = WMMA_BF16(aq0, b0, c);
        c = WMMA_BF16(aq1, b1, c);

#pragma unroll
        for (int r = 0; r < 8; ++r) {
            const int mm   = r + cm;
            const int nc   = wv * 16 + (ln & 15);     // 0..63 within tile
            const int ncol = kt * 64 + nc;
            S[mm * LL + ncol] = ms[mm * 64 + nc] ? c[r] * 0.125f : -10000.0f;
        }
    }
    __syncthreads();

    // ---- Phase 2: row softmax (keep unnormalized exp in S) ----
    {
        const int row = tid >> 3, ch = tid & 7;       // 8 threads/row, interleaved f4
        v4f* Srow4 = (v4f*)(S + row * LL);
        float mx = -3.0e38f;
        for (int j = 0; j < 32; ++j) {
            v4f v = Srow4[ch + 8 * j];
            mx = fmaxf(mx, fmaxf(fmaxf(v.x, v.y), fmaxf(v.z, v.w)));
        }
        red[row * 8 + ch] = mx;
        __syncthreads();
        float rmx = -3.0e38f;
        for (int j = 0; j < 8; ++j) rmx = fmaxf(rmx, red[row * 8 + j]);
        __syncthreads();
        float s = 0.f;
        for (int j = 0; j < 32; ++j) {
            v4f v = Srow4[ch + 8 * j];
            v.x = __expf(v.x - rmx); v.y = __expf(v.y - rmx);
            v.z = __expf(v.z - rmx); v.w = __expf(v.w - rmx);
            Srow4[ch + 8 * j] = v;
            s += (v.x + v.y) + (v.z + v.w);
        }
        red[row * 8 + ch] = s;
        __syncthreads();
        if (ch == 0) {
            float t = 0.f;
            for (int j = 0; j < 8; ++j) t += red[row * 8 + j];
            rsum[row] = t;
        }
    }
    __syncthreads();

    // ---- Phase 3: O = exp(S) @ V, row-normalized at store ----
    v8f acc = {};
    for (int ks = 0; ks < LL / 32; ++ks) {
        __syncthreads();
        const float* vg = V + (headoff + ks * 32) * DD;
        const v4f* vg4 = (const v4f*)vg;
        for (int i = tid; i < 32 * DD / 4; i += 128) {
            v4f v = vg4[i];
            v4bf w; w[0] = f2bf(v.x); w[1] = f2bf(v.y); w[2] = f2bf(v.z); w[3] = f2bf(v.w);
            ((v4bf*)kv)[i] = w;
        }
        if (ks + 1 < LL / 32) __builtin_prefetch(vg + 32 * DD, 0, 0);
        __syncthreads();

        v16bf ap, bv;
#pragma unroll
        for (int e = 0; e < 16; ++e) {
            const int k = koff + e;
            ap[e] = f2bf(S[am * LL + ks * 32 + k]);
            bv[e] = kv[(koff + e) * DD + wv * 16 + bn];
        }
        acc = WMMA_BF16(ap, bv, acc);
    }

#pragma unroll
    for (int r = 0; r < 8; ++r) {
        const int mm = r + cm;
        const int dc = wv * 16 + (ln & 15);
        const float inv = __builtin_amdgcn_rcpf(rsum[mm]);
        out[(headoff + q0 + mm) * DD + dc] = acc[r] * inv;
    }
}

__global__ void sdpa_init_kernel(float* ws) {
    if (threadIdx.x == 0 && blockIdx.x == 0) ws[0] = 0.f;
}

__global__ void sdpa_finalize_kernel(const float* __restrict__ ws,
                                     float* __restrict__ out_loss) {
    if (threadIdx.x == 0 && blockIdx.x == 0) {
        const float denom = (float)BB * (float)LL * (float)LL;   // B*L*L
        const float mse = ws[0] / denom;
        out_loss[0] = sqrtf(mse) / denom;
    }
}

extern "C" void kernel_launch(void* const* d_in, const int* in_sizes, int n_in,
                              void* d_out, int out_size, void* d_ws, size_t ws_size,
                              hipStream_t stream) {
    const float* Q    = (const float*)d_in[0];
    const float* K    = (const float*)d_in[1];
    const float* V    = (const float*)d_in[2];
    const int*   mask = (const int*)d_in[3];
    const float* bias = (const float*)d_in[4];
    float* out = (float*)d_out;
    float* acc = (float*)d_ws;

    sdpa_init_kernel<<<1, 32, 0, stream>>>(acc);

    dim3 lgrid(LL / 64, LL / 16, BB);          // (16, 64, 4)
    sdpa_loss_kernel<<<lgrid, 128, 0, stream>>>(Q, K, mask, bias, acc);

    dim3 agrid(LL / 16, HH, BB);               // (64, 16, 4)
    const size_t shbytes = (size_t)16 * LL * 4 + 64 * DD * 2 + 16 * 64 * 4
                         + 16 * 8 * 4 + 16 * 4;
    sdpa_attn_kernel<<<agrid, 128, shbytes, stream>>>(Q, K, V, mask, out);

    sdpa_finalize_kernel<<<1, 32, 0, stream>>>(
        acc, out + (size_t)BB * HH * LL * DD);
}
